// AbstractLayer_58849641889999
// MI455X (gfx1250) — matrix operand order repaired and checked by reading, not compiled, not run on hardware
//
#include <hip/hip_runtime.h>
#include <math.h>

#define N_PATH      8
#define D_IN        128
#define C_PER_PATH  128      // 2*BASE_OUT channels per path
#define BASE_OUT    64
#define VBS         256
#define BN_EPS      1e-5f

// LDS layout (bytes)
#define XS_STRIDE   136      // bf16 elems per x row (128 + 8 pad)
#define WS_STRIDE   136      // bf16 elems per weight row
#define YS_STRIDE   132      // f32 elems per y row (128 + 4 pad)
#define OFF_XS      0                    // 256*136*2 = 69632
#define OFF_W0      69632                // 128*136*2 = 34816
#define OFF_W1      104448               // 128*136*2 = 34816
#define OFF_YS      139264               // 256*132*4 = 135168
#define OFF_RS      274432               // 8*128*4   = 4096
#define OFF_RQ      278528               // 8*128*4   = 4096
#define OFF_SCL     282624               // 512
#define OFF_SFT     283136               // 512
#define SMEM_BYTES  283648               // <= 320 KB WGP LDS

typedef __attribute__((ext_vector_type(16))) __bf16        v16bf;
typedef __attribute__((ext_vector_type(8)))  float         v8f;
typedef __attribute__((ext_vector_type(4)))  unsigned int  v4u;
typedef __attribute__((ext_vector_type(4)))  float         v4f;

union BF16Frag { v4u u[2]; v16bf v; };

// Types matching the probed builtin prototype:
//   __builtin_amdgcn_global_load_async_to_lds_b128(int4 AS1*, int4 AS3*, Ii, Ii)
typedef int v4i_gcc __attribute__((vector_size(16)));
typedef __attribute__((address_space(1))) v4i_gcc* GlobalV4;
typedef __attribute__((address_space(3))) v4i_gcc* LdsV4;

__device__ __forceinline__ unsigned short f32_to_bf16_rne(float f) {
  unsigned int u = __float_as_uint(f);
  unsigned int r = u + 0x7FFFu + ((u >> 16) & 1u);
  return (unsigned short)(r >> 16);
}

// Issue 4 async global->LDS b128 loads per thread (k 0..63 of each row of one 128x128 bf16 weight
// tile per workgroup, padded rows). Tracked on ASYNCcnt, no VGPR data path.
__device__ __forceinline__ void stage_weights_async(const unsigned short* __restrict__ wsrc,
                                                    unsigned short* wdst_lds, int t) {
#pragma unroll
  for (int j = 0; j < 4; ++j) {
    int i  = t + j * 256;           // 0..1023 chunks of 8 bf16 (16 B)
    int o  = i >> 3;
    int k8 = (i & 7) << 3;
    const void* g = (const void*)(wsrc + o * D_IN + k8);
    void*       l = (void*)(wdst_lds + o * WS_STRIDE + k8);
    __builtin_amdgcn_global_load_async_to_lds_b128(
        (GlobalV4)(unsigned long long)(size_t)g,
        (LdsV4)(unsigned int)(size_t)l,   // low 32 bits of generic LDS addr == LDS offset
        0, 0);
  }
}

// ---------------------------------------------------------------------------
// Kernel 1: exact entmax15 (alpha=1.5) of mask_w [8][128] -> mask_f [8][128]
// ---------------------------------------------------------------------------
__global__ void entmax15_mask_kernel(const float* __restrict__ mask_w,
                                     float* __restrict__ mask_out) {
  __shared__ float xb[N_PATH][D_IN];    // (x - max)/2
  __shared__ float srt[N_PATH][D_IN];   // sorted descending
  __shared__ float tarr[N_PATH][D_IN];  // tau candidates
  __shared__ float taustar[N_PATH];
  const int t = threadIdx.x;            // blockDim = 128

  for (int i = t; i < N_PATH * D_IN; i += blockDim.x)
    xb[i / D_IN][i % D_IN] = mask_w[i];
  __syncthreads();

  if (t < N_PATH) {
    float mx = -3.0e38f;
    for (int i = 0; i < D_IN; ++i) mx = fmaxf(mx, xb[t][i]);
    for (int i = 0; i < D_IN; ++i) {
      float v = (xb[t][i] - mx) * 0.5f;
      xb[t][i] = v; srt[t][i] = v;
    }
    for (int i = 1; i < D_IN; ++i) {            // insertion sort, descending
      float key = srt[t][i];
      int j = i - 1;
      while (j >= 0 && srt[t][j] < key) { srt[t][j + 1] = srt[t][j]; --j; }
      srt[t][j + 1] = key;
    }
    float cs = 0.0f, cs2 = 0.0f;
    int support = 0;
    for (int j = 0; j < D_IN; ++j) {
      float v = srt[t][j];
      cs += v; cs2 += v * v;
      float rho    = (float)(j + 1);
      float mean   = cs  / rho;
      float meansq = cs2 / rho;
      float ss     = rho * (meansq - mean * mean);
      float delta  = fmaxf((1.0f - ss) / rho, 0.0f);
      float tau    = mean - sqrtf(delta);
      tarr[t][j]   = tau;
      support += (tau <= v) ? 1 : 0;
    }
    taustar[t] = tarr[t][support - 1];
  }
  __syncthreads();

  for (int i = t; i < N_PATH * D_IN; i += blockDim.x) {
    int p = i / D_IN;
    float v = fmaxf(xb[p][i % D_IN] - taustar[p], 0.0f);
    mask_out[i] = v * v;
  }
}

// ---------------------------------------------------------------------------
// Kernel 2: fold mask into conv_w, quantize to bf16 (conv_b cancels in BN).
// ---------------------------------------------------------------------------
__global__ void fold_weights_kernel(const float* __restrict__ conv_w,
                                    const float* __restrict__ mask,
                                    unsigned short* __restrict__ wfold) {
  int idx = blockIdx.x * blockDim.x + threadIdx.x;   // n*16384 + o*128 + k
  if (idx < N_PATH * C_PER_PATH * D_IN) {
    int k = idx & (D_IN - 1);
    int n = idx >> 14;
    wfold[idx] = f32_to_bf16_rne(conv_w[idx] * mask[n * D_IN + k]);
  }
}

// ---------------------------------------------------------------------------
// Kernel 3: fused per-virtual-batch pipeline, one WG per 256 rows.
// Double-buffered async-to-LDS weight staging overlapped with WMMA GEMM,
// ghost-BN, sigmoid-GLU, ReLU, path-sum in registers.
// ---------------------------------------------------------------------------
__global__ __launch_bounds__(256, 1)
void fused_tabnet_block_kernel(const float* __restrict__ x,
                               const unsigned short* __restrict__ wfold,
                               const float* __restrict__ gamma,
                               const float* __restrict__ beta,
                               float* __restrict__ out) {
  extern __shared__ __align__(16) char smem[];
  unsigned short* xs   = (unsigned short*)(smem + OFF_XS);   // [256][136] bf16
  unsigned short* wsm0 = (unsigned short*)(smem + OFF_W0);   // [128][136] bf16
  unsigned short* wsm1 = (unsigned short*)(smem + OFF_W1);   // [128][136] bf16
  float* ys    = (float*)(smem + OFF_YS);                    // [256][132] f32
  float* red_s = (float*)(smem + OFF_RS);                    // [8][128]
  float* red_q = (float*)(smem + OFF_RQ);                    // [8][128]
  float* scl   = (float*)(smem + OFF_SCL);                   // [128]
  float* sft   = (float*)(smem + OFF_SFT);                   // [128]

  const int t    = threadIdx.x;
  const int lane = t & 31;
  const int wave = t >> 5;
  const int row0 = blockIdx.x * VBS;

  // kick off async staging of path 0 weights while we convert the x tile
  stage_weights_async(wfold, wsm0, t);

  // ---- stage x tile: 256x128 f32 -> bf16 LDS (coalesced b128 loads) ----
  for (int i = t; i < VBS * (D_IN / 4); i += 256) {
    int r  = i >> 5;
    int c4 = (i & 31) << 2;
    v4f xv = *(const v4f*)(x + (size_t)(row0 + r) * D_IN + c4);
    unsigned long long p =
          (unsigned long long)f32_to_bf16_rne(xv[0])
        | ((unsigned long long)f32_to_bf16_rne(xv[1]) << 16)
        | ((unsigned long long)f32_to_bf16_rne(xv[2]) << 32)
        | ((unsigned long long)f32_to_bf16_rne(xv[3]) << 48);
    *(unsigned long long*)(xs + r * XS_STRIDE + c4) = p;
  }

  float acc[BASE_OUT];
#pragma unroll
  for (int o = 0; o < BASE_OUT; ++o) acc[o] = 0.0f;

  for (int p = 0; p < N_PATH; ++p) {
    // ---- overlap: issue async staging of NEXT path's weights, then wait
    //      for THIS path's staging (async loads complete in order) ----
    if (p + 1 < N_PATH) {
      stage_weights_async(wfold + (size_t)(p + 1) * (C_PER_PATH * D_IN),
                          (p & 1) ? wsm0 : wsm1, t);
      asm volatile("s_wait_asynccnt 0x4" ::: "memory");   // this path's 4 done
    } else {
      asm volatile("s_wait_asynccnt 0x0" ::: "memory");
    }
    __syncthreads();

    const unsigned short* wb = (p & 1) ? wsm1 : wsm0;

    // ---- WMMA GEMM: ys[256][128] = xs * wb^T ----
    // wave w owns rows [32w, 32w+32). A frag: lane%16 = row, K-halves by
    // lane/16; hoisted across the ni loop. B frag: lane%16 = out channel.
    const int mbase0 = wave * 32;
#pragma unroll
    for (int mi = 0; mi < 2; ++mi) {
      const int mbase = mbase0 + mi * 16;
      const unsigned short* aptr =
          xs + (mbase + (lane & 15)) * XS_STRIDE + ((lane < 16) ? 0 : 8);
      BF16Frag afr[4];
#pragma unroll
      for (int kb = 0; kb < 4; ++kb) {
        afr[kb].u[0] = *(const v4u*)(aptr + kb * 32);        // K = base+[0..7]
        afr[kb].u[1] = *(const v4u*)(aptr + kb * 32 + 16);   // K = base+16+[0..7]
      }
#pragma unroll
      for (int ni = 0; ni < 8; ++ni) {
        const unsigned short* bptr =
            wb + (ni * 16 + (lane & 15)) * WS_STRIDE + ((lane < 16) ? 0 : 8);
        v8f c = {};
#pragma unroll
        for (int kb = 0; kb < 4; ++kb) {
          BF16Frag bf;
          bf.u[0] = *(const v4u*)(bptr + kb * 32);       // K = base+[0..7]
          bf.u[1] = *(const v4u*)(bptr + kb * 32 + 16);  // K = base+16+[0..7]
          c = __builtin_amdgcn_wmma_f32_16x16x32_bf16(
                  false, afr[kb].v, false, bf.v, (short)0, c, false, false);
        }
        // D layout: VGPR r -> M = r + 8*(lane>=16), N = lane%16
        const int mrow = mbase + ((lane < 16) ? 0 : 8);
        const int ncol = ni * 16 + (lane & 15);
#pragma unroll
        for (int r = 0; r < 8; ++r)
          ys[(mrow + r) * YS_STRIDE + ncol] = c[r];
      }
    }
    __syncthreads();

    // ---- ghost-BN stats: vectorized b128 row reads, 8x32-row partials ----
    {
      const int cg = t & 31;           // channel group: channels 4*cg..4*cg+3
      const int rg = t >> 5;           // row group: rows 32*rg..32*rg+31
      v4f s = {}, q = {};
      for (int r = 0; r < 32; ++r) {
        v4f y4 = *(const v4f*)(ys + (rg * 32 + r) * YS_STRIDE + cg * 4);
        s += y4; q += y4 * y4;
      }
#pragma unroll
      for (int j = 0; j < 4; ++j) {
        red_s[rg * 128 + cg * 4 + j] = s[j];
        red_q[rg * 128 + cg * 4 + j] = q[j];
      }
    }
    __syncthreads();
    if (t < C_PER_PATH) {
      float s = 0.0f, q = 0.0f;
#pragma unroll
      for (int g = 0; g < 8; ++g) { s += red_s[g * 128 + t]; q += red_q[g * 128 + t]; }
      float mu = s * (1.0f / 256.0f);
      float vr = q * (1.0f / 256.0f) - mu * mu;   // biased var, matches reference
      float gm = gamma[p * C_PER_PATH + t];
      float bt = beta[p * C_PER_PATH + t];
      float sc = gm * rsqrtf(vr + BN_EPS);
      scl[t] = sc;
      sft[t] = bt - mu * sc;          // conv_b provably cancels inside BN
    }
    __syncthreads();

    // ---- normalize + sigmoid-GLU + ReLU, accumulate over paths ----
    {
      const float* yr = ys + t * YS_STRIDE;       // thread t owns row t
#pragma unroll
      for (int o = 0; o < BASE_OUT; ++o) {
        float ya = yr[o]      * scl[o]      + sft[o];
        float yb = yr[o + 64] * scl[o + 64] + sft[o + 64];
        float gl = yb / (1.0f + __expf(-ya));     // sigmoid(ya)*yb
        acc[o] += fmaxf(gl, 0.0f);
      }
    }
    __syncthreads();   // ys reused next path
  }

  // ---- write out [256][64] ----
  float* orow = out + (size_t)(row0 + t) * BASE_OUT;
#pragma unroll
  for (int o = 0; o < BASE_OUT; o += 4) {
    v4f v; v[0] = acc[o]; v[1] = acc[o + 1]; v[2] = acc[o + 2]; v[3] = acc[o + 3];
    *(v4f*)(orow + o) = v;
  }
}

// ---------------------------------------------------------------------------
extern "C" void kernel_launch(void* const* d_in, const int* in_sizes, int n_in,
                              void* d_out, int out_size, void* d_ws, size_t ws_size,
                              hipStream_t stream) {
  const float* x      = (const float*)d_in[0];
  const float* mask_w = (const float*)d_in[1];
  const float* conv_w = (const float*)d_in[2];
  // d_in[3] = conv_b: per-channel constant within each virtual batch -> exactly
  // cancelled by ghost-BN mean subtraction; intentionally unused.
  const float* gamma  = (const float*)d_in[4];
  const float* beta   = (const float*)d_in[5];
  float* out = (float*)d_out;

  const int B = in_sizes[0] / D_IN;          // 65536
  const int n_vb = B / VBS;                  // 256 workgroups

  float* mask_f = (float*)d_ws;                                    // 4 KB
  unsigned short* wfold = (unsigned short*)((char*)d_ws + 4096);   // 256 KB

  entmax15_mask_kernel<<<1, 128, 0, stream>>>(mask_w, mask_f);

  const int wtot = N_PATH * C_PER_PATH * D_IN;
  fold_weights_kernel<<<(wtot + 255) / 256, 256, 0, stream>>>(conv_w, mask_f, wfold);

  (void)hipFuncSetAttribute(reinterpret_cast<const void*>(fused_tabnet_block_kernel),
                            hipFuncAttributeMaxDynamicSharedMemorySize, SMEM_BYTES);
  fused_tabnet_block_kernel<<<n_vb, 256, SMEM_BYTES, stream>>>(x, wfold, gamma, beta, out);
}
